// AdditiveAttention_33423435497548
// MI455X (gfx1250) — matrix operand order repaired and checked by reading, not compiled, not run on hardware
//
#include <hip/hip_runtime.h>
#include <math.h>
#include <stdint.h>

// ---------------------------------------------------------------------------
// Types for CDNA5 WMMA (gfx1250, wave32)
// ---------------------------------------------------------------------------
typedef __bf16 bf16_t;
typedef __attribute__((ext_vector_type(16))) __bf16 v16bf;
typedef __attribute__((ext_vector_type(8)))  __bf16 v8bf;
typedef __attribute__((ext_vector_type(8)))  float  v8f;
typedef __attribute__((ext_vector_type(4)))  int    v4i;

#define B_    8
#define S_    2048
#define D_    1024
#define H_    16
#define M_TOT (B_ * S_)      /* 16384 rows */
#define SCALE_ 0.125f        /* 1/sqrt(64) */

// ---------------------------------------------------------------------------
// Async global->LDS plumbing (CDNA5 ASYNCcnt path), guarded so we fall back
// to plain load + ds_store if the builtin is absent on this toolchain.
// Probe-confirmed signature: (v4i AS1*, v4i AS3*, imm offset, imm cpol)
// ---------------------------------------------------------------------------
#ifdef __has_builtin
# if __has_builtin(__builtin_amdgcn_global_load_async_to_lds_b128)
#  define ASYNC_LDS 1
# endif
#endif

#ifdef ASYNC_LDS
# if __has_builtin(__builtin_amdgcn_s_wait_asynccnt)
#  define WAIT_ASYNC(n) __builtin_amdgcn_s_wait_asynccnt(n)
# else
#  define WAIT_ASYNC(n) asm volatile("s_wait_asynccnt " #n ::: "memory")
# endif
#else
# define WAIT_ASYNC(n) ((void)0)
#endif

typedef __attribute__((address_space(1))) void as1_void;
typedef __attribute__((address_space(3))) void as3_void;
typedef __attribute__((address_space(1))) v4i  as1_v4i;
typedef __attribute__((address_space(3))) v4i  as3_v4i;

__device__ __forceinline__ void g2l_b128(const bf16_t* g, bf16_t* l) {
#ifdef ASYNC_LDS
  as1_void* gas = (as1_void*)(void*)const_cast<bf16_t*>(g);
  as3_void* las = (as3_void*)(void*)l;
  __builtin_amdgcn_global_load_async_to_lds_b128(
      (as1_v4i*)gas, (as3_v4i*)las, 0, 0);
#else
  *(v8bf*)l = *(const v8bf*)g;
#endif
}

// ---------------------------------------------------------------------------
// Tiled bf16 WMMA GEMM:  C[M,N] = A[M,K] @ Bw[K,N] + bias[N]  (+ resid[M,N])
// Block: 256 threads (8 waves), tile 128x128, K-step 32, double-buffered LDS.
// Wave tile 32x64 -> 8 accumulators of v_wmma_f32_16x16x32_bf16.
// ---------------------------------------------------------------------------
#define MT 128
#define NT 128
#define KT 32
#define ASTR 40    /* LDS row stride for A tile (bf16): conflict-free, 16B mult */
#define BSTR 136   /* LDS row stride for B tile (bf16) */

__global__ __launch_bounds__(256) void gemm_bf16_kernel(
    const bf16_t* __restrict__ A,     // [M,K] row-major
    const bf16_t* __restrict__ Bw,    // [K,N] row-major
    const float*  __restrict__ bias,  // [N]
    const float*  __restrict__ resid, // [M,N] or nullptr
    float* __restrict__ C,            // [M,N]
    int M, int N, int K)
{
  __shared__ alignas(16) bf16_t Asm[2][MT * ASTR];
  __shared__ alignas(16) bf16_t Bsm[2][KT * BSTR];

  const int tid  = threadIdx.x;
  const int wave = tid >> 5;
  const int lane = tid & 31;
  const int m0 = blockIdx.x * MT;
  const int n0 = blockIdx.y * NT;
  const int wm = (wave >> 1) * 32;   // 0,32,64,96
  const int wn = (wave & 1) * 64;    // 0,64

  auto stage = [&](int buf, int k0) {
#pragma unroll
    for (int t = 0; t < 2; ++t) {
      const int c  = tid + t * 256;          // 512 chunks of 16B per tile
      const int ar = c >> 2;                 // 0..127
      const int ac = (c & 3) * 8;            // 0,8,16,24
      g2l_b128(A + (size_t)(m0 + ar) * K + (k0 + ac), &Asm[buf][ar * ASTR + ac]);
      const int br = c >> 4;                 // 0..31
      const int bc = (c & 15) * 8;           // 0..120
      g2l_b128(Bw + (size_t)(k0 + br) * N + (n0 + bc), &Bsm[buf][br * BSTR + bc]);
    }
  };

  v8f acc[2][4];
#pragma unroll
  for (int i = 0; i < 2; ++i)
#pragma unroll
    for (int j = 0; j < 4; ++j)
#pragma unroll
      for (int r = 0; r < 8; ++r) acc[i][j][r] = 0.0f;

  stage(0, 0);

  const int nk = K / KT;
  const int kb = (lane < 16) ? 0 : 8;   // A-frag K base per half-wave
  const int am = lane & 15;             // A-frag row within 16

  for (int kk = 0; kk < nk; ++kk) {
    const int buf = kk & 1;
    if (kk + 1 < nk) {
      stage(buf ^ 1, (kk + 1) * KT);   // prefetch next tile (4 async ops/thread)
      WAIT_ASYNC(4);                   // current tile's loads complete
    } else {
      WAIT_ASYNC(0);
    }
    __syncthreads();

    v16bf afrag[2], bfrag[4];
#pragma unroll
    for (int i = 0; i < 2; ++i) {
      // 16-bit A 16x32 layout: lane m=lane&15; K = kb + e (e<8), kb+16+(e-8)
      const bf16_t* p = &Asm[buf][(wm + i * 16 + am) * ASTR + kb];
      v8bf lo = *(const v8bf*)p;
      v8bf hi = *(const v8bf*)(p + 16);
      afrag[i] = __builtin_shufflevector(lo, hi,
          0,1,2,3,4,5,6,7,8,9,10,11,12,13,14,15);
    }
#pragma unroll
    for (int j = 0; j < 4; ++j) {
      // B 32x16 layout: lane n=lane&15; K = kb + e (e<8), kb+16+(e-8)
      const bf16_t* p = &Bsm[buf][kb * BSTR + wn + j * 16 + am];
#pragma unroll
      for (int e = 0; e < 8; ++e) {
        bfrag[j][e]     = p[e * BSTR];
        bfrag[j][e + 8] = p[(16 + e) * BSTR];
      }
    }
#pragma unroll
    for (int i = 0; i < 2; ++i)
#pragma unroll
      for (int j = 0; j < 4; ++j)
        acc[i][j] = __builtin_amdgcn_wmma_f32_16x16x32_bf16(
            false, afrag[i], false, bfrag[j], (short)0, acc[i][j], false, false);

    __syncthreads();  // protect current buf from next iteration's prefetch
  }

  // Epilogue: C frag layout -> VGPR r is row (r | r+8), lane&15 is col.
  const int rbase = m0 + wm + ((lane >> 4) << 3);
#pragma unroll
  for (int i = 0; i < 2; ++i) {
#pragma unroll
    for (int j = 0; j < 4; ++j) {
      const int col = n0 + wn + j * 16 + (lane & 15);
      const float bv = bias[col];
#pragma unroll
      for (int r = 0; r < 8; ++r) {
        const int row = rbase + i * 16 + r;
        const size_t idx = (size_t)row * N + col;
        float v = acc[i][j][r] + bv;
        if (resid) v += resid[idx];
        C[idx] = v;
      }
    }
  }
}

// ---------------------------------------------------------------------------
// Additive-attention pooling: one wave per row.
//   val[d] = Rm[row][d] * (scale ? scale[b][d] : 1)
//   logits = (val @ Ws + bs) * SCALE; w = softmax_H(logits)
//   gout[b][d] += w[d/64] * val[d]      (block LDS partial, then global atomic)
// ---------------------------------------------------------------------------
__global__ __launch_bounds__(256) void pool_kernel(
    const float* __restrict__ Rm,     // [M_TOT, D]
    const float* __restrict__ scale,  // [B, D] or nullptr
    const float* __restrict__ Ws,     // [D, H]
    const float* __restrict__ bs,     // [H]
    float* __restrict__ gout)         // [B, D] (pre-zeroed)
{
  __shared__ float accsh[D_];
  const int tid  = threadIdx.x;
  const int wave = tid >> 5;
  const int lane = tid & 31;
  const int row  = blockIdx.x * 8 + wave;   // 8 rows/block, never straddles batch
  const int b    = row >> 11;               // S = 2048

  for (int i = tid; i < D_; i += 256) accsh[i] = 0.0f;
  __syncthreads();

  const float* rp = Rm + (size_t)row * D_;
  const float* sp = scale ? (scale + (size_t)b * D_) : nullptr;

  float v[32];
#pragma unroll
  for (int c = 0; c < 32; ++c) {
    const int d = c * 32 + lane;
    float x = rp[d];
    if (sp) x *= sp[d];
    v[c] = x;
  }

  float part[H_];
#pragma unroll
  for (int h = 0; h < H_; ++h) part[h] = 0.0f;
#pragma unroll
  for (int c = 0; c < 32; ++c) {
    const int d = c * 32 + lane;
    const float* wr = Ws + (size_t)d * H_;
#pragma unroll
    for (int h = 0; h < H_; ++h) part[h] = fmaf(v[c], wr[h], part[h]);
  }
  // wave32 cross-lane reduction
#pragma unroll
  for (int h = 0; h < H_; ++h) {
#pragma unroll
    for (int off = 16; off > 0; off >>= 1)
      part[h] += __shfl_xor(part[h], off, 32);
  }

  float lg[H_];
  float mx = -1e30f;
#pragma unroll
  for (int h = 0; h < H_; ++h) {
    lg[h] = (part[h] + bs[h]) * SCALE_;
    mx = fmaxf(mx, lg[h]);
  }
  float den = 0.0f;
#pragma unroll
  for (int h = 0; h < H_; ++h) {
    lg[h] = __expf(lg[h] - mx);
    den += lg[h];
  }
  const float inv = 1.0f / den;

#pragma unroll
  for (int c = 0; c < 32; ++c) {
    const int d = c * 32 + lane;
    atomicAdd(&accsh[d], lg[d >> 6] * inv * v[c]);  // head = d / 64
  }
  __syncthreads();
  for (int i = tid; i < D_; i += 256)
    atomicAdd(&gout[(size_t)b * D_ + i], accsh[i]);
}

// ---------------------------------------------------------------------------
// Small elementwise kernels
// ---------------------------------------------------------------------------
__global__ void cvt_f32_to_bf16(const float* __restrict__ in,
                                bf16_t* __restrict__ out, int n) {
  int i = blockIdx.x * blockDim.x + threadIdx.x;
  const int stride = gridDim.x * blockDim.x;
  for (; i < n; i += stride) out[i] = (bf16_t)in[i];
}

__global__ void zero_f32(float* __restrict__ p, int n) {
  const int i = blockIdx.x * blockDim.x + threadIdx.x;
  if (i < n) p[i] = 0.0f;
}

__global__ void make_u(const float* __restrict__ q,
                       const float* __restrict__ gk,
                       bf16_t* __restrict__ u) {
  const int i = blockIdx.x * blockDim.x + threadIdx.x; // total = M_TOT*D_
  const int d = i & (D_ - 1);
  const int b = i >> 21;                               // / (S_*D_)
  u[i] = (bf16_t)(q[i] * gk[b * D_ + d]);
}

// ---------------------------------------------------------------------------
// Launcher
// ---------------------------------------------------------------------------
extern "C" void kernel_launch(void* const* d_in, const int* in_sizes, int n_in,
                              void* d_out, int out_size, void* d_ws, size_t ws_size,
                              hipStream_t stream) {
  const float* X   = (const float*)d_in[0];
  const float* Wqv = (const float*)d_in[1];
  const float* bqv = (const float*)d_in[2];
  const float* Wk  = (const float*)d_in[3];
  const float* bk  = (const float*)d_in[4];
  const float* Wqs = (const float*)d_in[5];
  const float* bqs = (const float*)d_in[6];
  const float* Wks = (const float*)d_in[7];
  const float* bks = (const float*)d_in[8];
  const float* Wr  = (const float*)d_in[9];
  const float* br  = (const float*)d_in[10];
  float* out = (float*)d_out;

  // Workspace carve-out (all chunks 256B-aligned by construction)
  char* p = (char*)d_ws;
  bf16_t* Xb   = (bf16_t*)p; p += (size_t)M_TOT * D_ * sizeof(bf16_t);
  bf16_t* Wqvb = (bf16_t*)p; p += (size_t)D_ * D_ * sizeof(bf16_t);
  bf16_t* Wkb  = (bf16_t*)p; p += (size_t)D_ * D_ * sizeof(bf16_t);
  bf16_t* Wrb  = (bf16_t*)p; p += (size_t)D_ * D_ * sizeof(bf16_t);
  bf16_t* Ub   = (bf16_t*)p; p += (size_t)M_TOT * D_ * sizeof(bf16_t);
  float*  qf   = (float*)p;  p += (size_t)M_TOT * D_ * sizeof(float);
  float*  kf   = (float*)p;  p += (size_t)M_TOT * D_ * sizeof(float);
  float*  gq   = (float*)p;  p += (size_t)B_ * D_ * sizeof(float);
  float*  gk   = (float*)p;  p += (size_t)B_ * D_ * sizeof(float);

  // 1) bf16 staging of X and the three big weight matrices
  cvt_f32_to_bf16<<<4096, 256, 0, stream>>>(X,   Xb,   M_TOT * D_);
  cvt_f32_to_bf16<<<1024, 256, 0, stream>>>(Wqv, Wqvb, D_ * D_);
  cvt_f32_to_bf16<<<1024, 256, 0, stream>>>(Wk,  Wkb,  D_ * D_);
  cvt_f32_to_bf16<<<1024, 256, 0, stream>>>(Wr,  Wrb,  D_ * D_);
  zero_f32<<<(2 * B_ * D_ + 255) / 256, 256, 0, stream>>>(gq, 2 * B_ * D_); // gq+gk

  const dim3 gg(M_TOT / MT, D_ / NT);  // 128 x 8 blocks

  // 2) q = X@Wqv + bqv ; k = X@Wk + bk
  gemm_bf16_kernel<<<gg, 256, 0, stream>>>(Xb, Wqvb, bqv, nullptr, qf, M_TOT, D_, D_);
  gemm_bf16_kernel<<<gg, 256, 0, stream>>>(Xb, Wkb,  bk,  nullptr, kf, M_TOT, D_, D_);

  // 3) global_query from q ; global_key from p = k * gq
  pool_kernel<<<M_TOT / 8, 256, 0, stream>>>(qf, nullptr, Wqs, bqs, gq);
  pool_kernel<<<M_TOT / 8, 256, 0, stream>>>(kf, gq,      Wks, bks, gk);

  // 4) u = q * gk (v shares q), then out = q + u@Wr + br
  make_u<<<(M_TOT * D_) / 256, 256, 0, stream>>>(qf, gk, Ub);
  gemm_bf16_kernel<<<gg, 256, 0, stream>>>(Ub, Wrb, br, qf, out, M_TOT, D_, D_);
}
